// QNetwork_10814727651980
// MI455X (gfx1250) — matrix operand order repaired and checked by reading, not compiled, not run on hardware
//
#include <hip/hip_runtime.h>
#include <cstddef>

#define Nn   6000
#define Pp   100
#define PPp  112
#define KPp  6016
#define GSs  200
#define Tt   4

typedef __attribute__((ext_vector_type(16))) __bf16 v16bf;
typedef __attribute__((ext_vector_type(8)))  float  v8f;

__device__ __forceinline__ unsigned short f2bf(float f) {
  unsigned int u = __float_as_uint(f);
  u += 0x7FFFu + ((u >> 16) & 1u);          // round-to-nearest-even
  return (unsigned short)(u >> 16);
}

// ---------------- weight norm: w[r,c] = v[r,c] * g[r] / ||v[r,:]|| ----------
__global__ void wnorm_k(const float* __restrict__ v, const float* __restrict__ g,
                        float* __restrict__ w, int R, int C) {
  __shared__ float red[64];
  int r = blockIdx.x;
  if (r >= R) return;
  float s = 0.f;
  for (int c = threadIdx.x; c < C; c += 64) { float x = v[r * C + c]; s += x * x; }
  red[threadIdx.x] = s;
  __syncthreads();
  for (int o = 32; o > 0; o >>= 1) {
    if (threadIdx.x < o) red[threadIdx.x] += red[threadIdx.x + o];
    __syncthreads();
  }
  float scale = g[r] / sqrtf(red[0]);
  for (int c = threadIdx.x; c < C; c += 64) w[r * C + c] = v[r * C + c] * scale;
}

// ---------------- per-row sums of A and edge features -----------------------
__global__ void row_sums_k(const float* __restrict__ A, const float* __restrict__ E,
                           float* __restrict__ rowsum, float* __restrict__ efsum) {
  __shared__ float ra[256], re[256];
  int i = blockIdx.x;
  const float* ar = A + (size_t)i * Nn;
  const float* er = E + (size_t)i * Nn;
  float sa = 0.f, se = 0.f;
  for (int j = threadIdx.x; j < Nn; j += 256) { sa += ar[j]; se += er[j]; }
  ra[threadIdx.x] = sa; re[threadIdx.x] = se;
  __syncthreads();
  for (int o = 128; o > 0; o >>= 1) {
    if (threadIdx.x < o) { ra[threadIdx.x] += ra[threadIdx.x + o];
                           re[threadIdx.x] += re[threadIdx.x + o]; }
    __syncthreads();
  }
  if (threadIdx.x == 0) { rowsum[i] = ra[0]; efsum[i] = re[0]; }
}

// ---------------- term1 + term3 (constant across iterations) ----------------
__global__ void term13_k(const float* __restrict__ nf, const float* __restrict__ efsum,
                         const float* __restrict__ w1, const float* __restrict__ w3,
                         const float* __restrict__ w4, float* __restrict__ out) {
  int idx = blockIdx.x * 256 + threadIdx.x;
  if (idx >= Nn * Pp) return;
  int i = idx / Pp, q = idx - i * Pp;
  float es  = efsum[i];
  float acc = nf[i] * w1[q];
  const float* w3r = w3 + q * Pp;
  for (int p = 0; p < Pp; ++p)
    acc += fmaxf(es * w4[p], 0.f) * w3r[p];
  out[idx] = acc;
}

// ---------------- A (f32 [N][N]) -> bf16 zero-padded [N][KPp], once ---------
__global__ void aconv_k(const float* __restrict__ A, unsigned int* __restrict__ Abf) {
  int idx = blockIdx.x * 256 + threadIdx.x;            // one uint = 2 bf16
  const int pairsPerRow = KPp / 2;                     // 3008
  if (idx >= Nn * pairsPerRow) return;
  int i  = idx / pairsPerRow;
  int k0 = (idx - i * pairsPerRow) * 2;
  unsigned int lo = 0, hi = 0;
  if (k0 < Nn) {                                       // Nn even: pair all-valid or all-pad
    const float* ar = A + (size_t)i * Nn + k0;
    lo = f2bf(ar[0]);
    hi = f2bf(ar[1]);
  }
  Abf[idx] = (hi << 16) | lo;
}

// ---------------- emb (f32, [N][P]) -> transposed padded bf16 [PPp][KPp] ----
__global__ void convert_bt_k(const float* __restrict__ src, unsigned short* __restrict__ BT) {
  int idx = blockIdx.x * 256 + threadIdx.x;
  if (idx >= PPp * KPp) return;
  int p = idx / KPp;
  int k = idx - p * KPp;
  float v = (p < Pp && k < Nn) ? src[(size_t)k * Pp + p] : 0.f;
  BT[idx] = f2bf(v);
}

// ---------------- big GEMM, pre-converted bf16 A path -----------------------
// one wave = 16 rows x 112 cols (7 tiles), K split 4 ways (deterministic partials)
__global__ __launch_bounds__(128) void gemm_pre_k(const unsigned short* __restrict__ Abf,
                                                  const unsigned short* __restrict__ BT,
                                                  float* __restrict__ Hp) {
  const int job  = blockIdx.x * 4 + (threadIdx.x >> 5);  // 375*4 = 1500 jobs
  const int ks   = job & 3;
  const int mt   = job >> 2;
  const int m0   = mt * 16;
  const int lane = threadIdx.x & 31;
  const int half = lane >> 4;
  const int mr   = lane & 15;

  const unsigned short* arow = Abf + (size_t)(m0 + mr) * KPp;
  v8f acc[7] = {};

  const int kb0 = ks * 1504;
  const int kb1 = kb0 + 1504;
  for (int kb = kb0; kb < kb1; kb += 32) {
    // A tile 16x32 bf16, layout per ISA 7.12.2: e<8 -> K=half*8+e ; e>=8 -> K=16+half*8+e-8
    union { v16bf v; unsigned int w[8]; } a;
    *(uint4*)(a.w)     = *(const uint4*)(arow + kb + half * 8);
    *(uint4*)(a.w + 4) = *(const uint4*)(arow + kb + 16 + half * 8);

#pragma unroll
    for (int t = 0; t < 7; ++t) {
      union { v16bf v; unsigned int w[8]; } b;
      const uint4* bp = (const uint4*)(BT + (size_t)(t * 16 + mr) * KPp + kb + half * 8);
      *(uint4*)(b.w)     = bp[0];
      *(uint4*)(b.w + 4) = bp[2];
      acc[t] = __builtin_amdgcn_wmma_f32_16x16x32_bf16(
          false, a.v, false, b.v, (short)0, acc[t], false, false);
    }
  }

  float* Hout = Hp + (size_t)ks * (size_t)Nn * PPp;
#pragma unroll
  for (int t = 0; t < 7; ++t)
#pragma unroll
    for (int v = 0; v < 8; ++v)
      Hout[(size_t)(m0 + half * 8 + v) * PPp + t * 16 + mr] = acc[t][v];
}

// ---------------- fallback GEMM: f32 A converted in-register ----------------
__global__ __launch_bounds__(128) void gemm_fly_k(const float* __restrict__ A,
                                                  const unsigned short* __restrict__ BT,
                                                  float* __restrict__ Hp) {
  const int job  = blockIdx.x * 4 + (threadIdx.x >> 5);
  const int ks   = job & 3;
  const int mt   = job >> 2;
  const int m0   = mt * 16;
  const int lane = threadIdx.x & 31;
  const int half = lane >> 4;
  const int mr   = lane & 15;

  const float* arow = A + (size_t)(m0 + mr) * Nn;
  v8f acc[7] = {};

  const int kb0 = ks * 1504;
  const int kb1 = kb0 + 1504;
  for (int kb = kb0; kb < kb1; kb += 32) {
    union { v16bf v; unsigned short u[16]; } a;
    const float* p1 = arow + kb + half * 8;
    float4 c10 = *(const float4*)(p1);
    float4 c11 = *(const float4*)(p1 + 4);
    float4 c20, c21;
    if (kb < 5984) {                 // tail step: cols 6000..6015 are zero
      c20 = *(const float4*)(p1 + 16);
      c21 = *(const float4*)(p1 + 20);
    } else {
      c20 = make_float4(0.f, 0.f, 0.f, 0.f);
      c21 = make_float4(0.f, 0.f, 0.f, 0.f);
    }
    a.u[0]  = f2bf(c10.x); a.u[1]  = f2bf(c10.y); a.u[2]  = f2bf(c10.z); a.u[3]  = f2bf(c10.w);
    a.u[4]  = f2bf(c11.x); a.u[5]  = f2bf(c11.y); a.u[6]  = f2bf(c11.z); a.u[7]  = f2bf(c11.w);
    a.u[8]  = f2bf(c20.x); a.u[9]  = f2bf(c20.y); a.u[10] = f2bf(c20.z); a.u[11] = f2bf(c20.w);
    a.u[12] = f2bf(c21.x); a.u[13] = f2bf(c21.y); a.u[14] = f2bf(c21.z); a.u[15] = f2bf(c21.w);

#pragma unroll
    for (int t = 0; t < 7; ++t) {
      union { v16bf v; unsigned int w[8]; } b;
      const uint4* bp = (const uint4*)(BT + (size_t)(t * 16 + mr) * KPp + kb + half * 8);
      *(uint4*)(b.w)     = bp[0];
      *(uint4*)(b.w + 4) = bp[2];
      acc[t] = __builtin_amdgcn_wmma_f32_16x16x32_bf16(
          false, a.v, false, b.v, (short)0, acc[t], false, false);
    }
  }

  float* Hout = Hp + (size_t)ks * (size_t)Nn * PPp;
#pragma unroll
  for (int t = 0; t < 7; ++t)
#pragma unroll
    for (int v = 0; v < 8; ++v)
      Hout[(size_t)(m0 + half * 8 + v) * PPp + t * 16 + mr] = acc[t][v];
}

// ---------------- sb[q] = sum_g state[g] * w8[q,g] --------------------------
__global__ void sb_k(const float* __restrict__ state, const float* __restrict__ w8,
                     float* __restrict__ sb) {
  int q = threadIdx.x;
  if (q >= Pp) return;
  float a = 0.f;
  const float* wr = w8 + q * GSs;
  for (int g = 0; g < GSs; ++g) a += state[g] * wr[g];
  sb[q] = a;
}

// ---------------- epilogue: emb_new = relu(t13 + (H@w2.T)/rowsum (+sb)) -----
__global__ void epilogue_k(const float* __restrict__ Hp, const float* __restrict__ t13,
                           const float* __restrict__ w2, const float* __restrict__ rowsum,
                           const float* __restrict__ sb, float* __restrict__ dst, int nv) {
  int idx = blockIdx.x * 256 + threadIdx.x;
  if (idx >= Nn * Pp) return;
  int i = idx / Pp, q = idx - i * Pp;
  const float* wr = w2 + q * Pp;
  const float* h0 = Hp + (size_t)i * PPp;
  const size_t seg = (size_t)Nn * PPp;
  float acc = 0.f;
  for (int p = 0; p < Pp; ++p) {
    float h = h0[p] + h0[seg + p] + h0[2 * seg + p] + h0[3 * seg + p];
    acc += h * wr[p];
  }
  acc /= rowsum[i];
  acc += t13[idx];
  if (i >= nv) acc += sb[q];
  dst[idx] = fmaxf(acc, 0.f);
}

// ---------------- state[g] = relu( (colmean of emb[nv:]) . w9[g,:] ) --------
__global__ void state_k(const float* __restrict__ emb, const float* __restrict__ w9,
                        float* __restrict__ state, int nv) {
  __shared__ float cm[Pp];
  int tid  = threadIdx.x;
  int rows = Nn - nv;
  if (tid < Pp) {
    float s = 0.f;
    for (int r = 0; r < rows; ++r) s += emb[(size_t)(nv + r) * Pp + tid];
    cm[tid] = s;
  }
  __syncthreads();
  if (tid < GSs) {
    float a = 0.f;
    const float* wr = w9 + tid * Pp;
    for (int p = 0; p < Pp; ++p) a += cm[p] * wr[p];
    state[tid] = fmaxf(a / (float)rows, 0.f);
  }
}

// ---------------- qa = (relu(state@w6.T)) . w51 -----------------------------
__global__ void qa_k(const float* __restrict__ state, const float* __restrict__ w6,
                     const float* __restrict__ w51, float* __restrict__ qa) {
  __shared__ float av[Pp];
  int tid = threadIdx.x;
  if (tid < Pp) {
    float a = 0.f;
    const float* wr = w6 + tid * GSs;
    for (int g = 0; g < GSs; ++g) a += state[g] * wr[g];
    av[tid] = fmaxf(a, 0.f);
  }
  __syncthreads();
  if (tid == 0) {
    float s = 0.f;
    for (int p = 0; p < Pp; ++p) s += av[p] * w51[p];
    qa[0] = s;
  }
}

// ---------------- q head: out[NP + c*U + u] = qa + relu(emb_u@w7.T).w52[c] --
__global__ void q_k(const float* __restrict__ emb, const int* __restrict__ uidx,
                    const float* __restrict__ w7, const float* __restrict__ w52,
                    const float* __restrict__ qa, float* __restrict__ out, int U) {
  __shared__ float b[Pp];
  int u   = blockIdx.x;
  int i   = uidx[u];
  int tid = threadIdx.x;
  if (tid < Pp) {
    float a = 0.f;
    const float* wr = w7 + tid * Pp;
    const float* er = emb + (size_t)i * Pp;
    for (int p = 0; p < Pp; ++p) a += er[p] * wr[p];
    b[tid] = fmaxf(a, 0.f);
  }
  __syncthreads();
  if (tid < 2) {
    float s = qa[0];
    const float* wr = w52 + tid * Pp;
    for (int p = 0; p < Pp; ++p) s += b[p] * wr[p];
    out[Nn * Pp + tid * U + u] = s;
  }
}

// ---------------- final emb + delta copy ------------------------------------
__global__ void out_k(const float* __restrict__ fin, const float* __restrict__ prev,
                      float* __restrict__ out, int U) {
  int idx = blockIdx.x * 256 + threadIdx.x;
  if (idx >= Nn * Pp) return;
  out[idx] = fin[idx];
  out[Nn * Pp + 2 * U + idx] = prev[idx] - fin[idx];
}

extern "C" void kernel_launch(void* const* d_in, const int* in_sizes, int n_in,
                              void* d_out, int out_size, void* d_ws, size_t ws_size,
                              hipStream_t stream) {
  (void)n_in; (void)out_size;
  const float* nf   = (const float*)d_in[0];
  const float* A    = (const float*)d_in[1];
  const float* E    = (const float*)d_in[2];
  const float* emb0 = (const float*)d_in[3];
  const int*   uidx = (const int*)d_in[4];
  const int U  = in_sizes[4];
  const int nv = in_sizes[4] + in_sizes[5];
  float* out = (float*)d_out;

  // ---- workspace carve (16B-aligned chunks) ----
  float* p = (float*)d_ws;
  auto carve = [&](size_t n) { float* r = p; p += ((n + 3) & ~(size_t)3); return r; };
  float* w1     = carve(Pp);
  float* w2     = carve((size_t)Pp * Pp);
  float* w3     = carve((size_t)Pp * Pp);
  float* w4     = carve(Pp);
  float* w51    = carve(Pp);
  float* w52    = carve(2 * Pp);
  float* w6     = carve((size_t)Pp * GSs);
  float* w7     = carve((size_t)Pp * Pp);
  float* w8     = carve((size_t)Pp * GSs);
  float* w9     = carve((size_t)GSs * Pp);
  float* rowsum = carve(Nn);
  float* efsum  = carve(Nn);
  float* t13    = carve((size_t)Nn * Pp);
  float* embA   = carve((size_t)Nn * Pp);
  float* embB   = carve((size_t)Nn * Pp);
  float* Hp     = carve((size_t)4 * Nn * PPp);
  unsigned short* BT = (unsigned short*)carve((size_t)PPp * KPp / 2);
  float* state  = carve(GSs);
  float* sb     = carve(Pp);
  float* qa     = carve(4);
  unsigned short* Abf = (unsigned short*)carve((size_t)Nn * KPp / 2);  // 72 MB, optional
  const size_t needed = (size_t)((char*)p - (char*)d_ws);
  const bool usePre = (needed <= ws_size);   // fixed per session -> deterministic

  // ---- weight normalization ----
  struct { int vi, gi; float* w; int R, C; } wn[10] = {
    { 6,  7, w1,  100,   1}, { 8,  9, w2,  100, 100}, {10, 11, w3,  100, 100},
    {12, 13, w4,  100,   1}, {14, 15, w51,   1, 100}, {16, 17, w52,   2, 100},
    {18, 19, w6,  100, 200}, {20, 21, w7,  100, 100}, {22, 23, w8,  100, 200},
    {24, 25, w9,  200, 100},
  };
  for (int i = 0; i < 10; ++i)
    wnorm_k<<<wn[i].R, 64, 0, stream>>>((const float*)d_in[wn[i].vi],
                                        (const float*)d_in[wn[i].gi],
                                        wn[i].w, wn[i].R, wn[i].C);

  row_sums_k<<<Nn, 256, 0, stream>>>(A, E, rowsum, efsum);
  term13_k<<<(Nn * Pp + 255) / 256, 256, 0, stream>>>(nf, efsum, w1, w3, w4, t13);
  state_k<<<1, 256, 0, stream>>>(emb0, w9, state, nv);
  if (usePre)
    aconv_k<<<(Nn * (KPp / 2) + 255) / 256, 256, 0, stream>>>(A, (unsigned int*)Abf);

  const float* src  = emb0;
  const float* prev = emb0;
  float* dst = embA;
  for (int t = 0; t < Tt; ++t) {
    convert_bt_k<<<(PPp * KPp + 255) / 256, 256, 0, stream>>>(src, BT);
    sb_k<<<1, 128, 0, stream>>>(state, w8, sb);
    if (usePre)
      gemm_pre_k<<<375, 128, 0, stream>>>(Abf, BT, Hp);
    else
      gemm_fly_k<<<375, 128, 0, stream>>>(A, BT, Hp);
    epilogue_k<<<(Nn * Pp + 255) / 256, 256, 0, stream>>>(Hp, t13, w2, rowsum, sb, dst, nv);
    state_k<<<1, 256, 0, stream>>>(dst, w9, state, nv);
    prev = src;
    src  = dst;
    dst  = (dst == embA) ? embB : embA;
  }
  // src = emb_T, prev = emb_{T-1}

  qa_k<<<1, 128, 0, stream>>>(state, w6, w51, qa);
  q_k<<<U, 128, 0, stream>>>(src, uidx, w7, w52, qa, out, U);
  out_k<<<(Nn * Pp + 255) / 256, 256, 0, stream>>>(src, prev, out, U);
}
